// DeepseekV4Attention_20675972563019
// MI455X (gfx1250) — matrix operand order repaired and checked by reading, not compiled, not run on hardware
//
#include <hip/hip_runtime.h>
#include <hip/hip_bf16.h>
#include <math.h>

typedef __attribute__((ext_vector_type(16))) _Float16 v16h;
typedef __attribute__((ext_vector_type(8)))  _Float16 v8h;
typedef __attribute__((ext_vector_type(8)))  float    v8f;
typedef __attribute__((ext_vector_type(4)))  float    f4;

#define S_LEN  2048
#define DHEAD  128
#define NHEAD  32
#define NKV    4
#define QTILE  16            // query rows per wave
#define WAVES  4             // waves per block
#define QBLK   (QTILE*WAVES) // 64 query rows per block
#define KBLK   64            // keys per iteration
#define KPAD   136           // Ksh row stride (f16 elems), 128+8 pad
#define VPAD   72            // VshT row stride, 64+8 pad
#define PPAD   72            // Psh row stride, 64+8 pad

#define LOG2E  1.4426950408889634f
// 1/sqrt(128) * log2(e): scores produced directly in log2 domain for v_exp_f32
#define LSCALE 0.12751512963530857f

union U16h { v16h v; _Float16 h[16]; };

__device__ __forceinline__ float fast_exp2(float x) {
#if defined(__HIP_DEVICE_COMPILE__) && __has_builtin(__builtin_amdgcn_exp2f)
    return __builtin_amdgcn_exp2f(x);
#else
    return exp2f(x);   // HIP provides a device overload; host pass parses fine
#endif
}

// ---------------------------------------------------------------------------
// Preprocess 1: K fp32 -> f16, same layout [NKV][S][D]
__global__ __launch_bounds__(256) void cvt_k_kernel(const float* __restrict__ src,
                                                    _Float16* __restrict__ dst) {
    const size_t i = ((size_t)blockIdx.x * 256 + threadIdx.x) * 8;
    f4 a = __builtin_nontemporal_load((const f4*)(src + i));
    f4 b = __builtin_nontemporal_load((const f4*)(src + i + 4));
    U16h o;  // only 8 lanes used
    o.h[0]=(_Float16)a.x; o.h[1]=(_Float16)a.y; o.h[2]=(_Float16)a.z; o.h[3]=(_Float16)a.w;
    o.h[4]=(_Float16)b.x; o.h[5]=(_Float16)b.y; o.h[6]=(_Float16)b.z; o.h[7]=(_Float16)b.w;
    __builtin_nontemporal_store(*(const v8h*)o.h, (v8h*)(dst + i));
}

// ---------------------------------------------------------------------------
// Preprocess 2: V fp32 [NKV][S][D] -> f16 transposed [NKV][D][S]
__global__ __launch_bounds__(256) void cvt_vT_kernel(const float* __restrict__ V,
                                                     _Float16* __restrict__ VT) {
    const int lin  = blockIdx.x * 256 + threadIdx.x;
    const int sg   = lin & (S_LEN/4 - 1);     // 512 groups of 4 along S (coalesced writes)
    const int rest = lin >> 9;
    const int d    = rest & (DHEAD - 1);
    const int kvh  = rest >> 7;
    const int s    = sg * 4;
    const float* base = V + ((size_t)kvh * S_LEN + s) * DHEAD + d;
    _Float16 o0 = (_Float16)__builtin_nontemporal_load(base);
    _Float16 o1 = (_Float16)__builtin_nontemporal_load(base + DHEAD);
    _Float16 o2 = (_Float16)__builtin_nontemporal_load(base + 2*DHEAD);
    _Float16 o3 = (_Float16)__builtin_nontemporal_load(base + 3*DHEAD);
    _Float16* out = VT + ((size_t)kvh * DHEAD + d) * S_LEN + s;
    out[0]=o0; out[1]=o1; out[2]=o2; out[3]=o3;
}

// ---------------------------------------------------------------------------
// Main flash-attention kernel with sink column folded into online softmax.
// All logits kept in log2 domain (scores pre-scaled by 1/sqrt(D)*log2(e)).
__global__ __launch_bounds__(128) void attn_sink_fa_kernel(
    const float* __restrict__ Q, const _Float16* __restrict__ K16,
    const _Float16* __restrict__ V16T, const float* __restrict__ sinks,
    float* __restrict__ Out)
{
    __shared__ __align__(16) _Float16 Ksh[KBLK * KPAD];          // [key][d]
    __shared__ __align__(16) _Float16 VshT[DHEAD * VPAD];        // [d][key]
    __shared__ __align__(16) _Float16 Psh[WAVES * QTILE * PPAD]; // per-wave [row][key]

    const int tid  = threadIdx.x;
    const int wave = tid >> 5;
    const int lane = tid & 31;
    const int lh   = lane >> 4;   // half of wave (0 or 1)
    const int ln   = lane & 15;

    const int bx    = blockIdx.x;
    const int head  = bx >> 5;            // / (S/QBLK = 32)
    const int qblk  = bx & 31;
    const int kvh   = head >> 3;          // 8 query heads per KV head
    const int qbase = qblk * QBLK + wave * QTILE;

    const float*    Qh = Q    + (size_t)head * S_LEN * DHEAD;
    const _Float16* Kh = K16  + (size_t)kvh  * S_LEN * DHEAD;
    const _Float16* Vh = V16T + (size_t)kvh  * DHEAD * S_LEN;

    // ---- Q tile -> WMMA A layout (4 chunks of 32 along D), convert once ----
    U16h aq[4];
    {
        const float* qrow = Qh + (size_t)(qbase + ln) * DHEAD;
        #pragma unroll
        for (int c = 0; c < 4; ++c) {
            #pragma unroll
            for (int half8 = 0; half8 < 2; ++half8) {
                const float* p = qrow + 32*c + lh*16 + half8*8;
                f4 a = *(const f4*)p;
                f4 b = *(const f4*)(p + 4);
                _Float16* h = aq[c].h + half8*8;
                h[0]=(_Float16)a.x; h[1]=(_Float16)a.y; h[2]=(_Float16)a.z; h[3]=(_Float16)a.w;
                h[4]=(_Float16)b.x; h[5]=(_Float16)b.y; h[6]=(_Float16)b.z; h[7]=(_Float16)b.w;
            }
        }
    }

    U16h ones;
    #pragma unroll
    for (int i = 0; i < 16; ++i) ones.h[i] = (_Float16)1.0f;

    // ---- state ----------------------------------------------------------
    v8f acc[8];
    #pragma unroll
    for (int t = 0; t < 8; ++t) acc[t] = (v8f){0,0,0,0,0,0,0,0};
    v8f sumacc = (v8f){0,0,0,0,0,0,0,0};  // running softmax row-sums (via ones-WMMA)
    float mrow[8];
    const float snk2 = sinks[head] * LOG2E;   // sink logit in log2 domain
    #pragma unroll
    for (int r = 0; r < 8; ++r) mrow[r] = snk2;

    const int nkb = qblk + 1;   // 64-key blocks up to causal bound (block-uniform)

    for (int kb = 0; kb < nkb; ++kb) {
        const int kbase = kb * KBLK;
        __syncthreads();   // previous iteration's LDS reads done

        // ---- stage K block: [64 keys][128 d] f16 (pure copy) -----------
        {
            const int kr  = tid >> 1;           // 0..63
            const int seg = (tid & 1) * 64;     // 0 or 64
            const _Float16* src = Kh + (size_t)(kbase + kr) * DHEAD + seg;
            _Float16* dst = &Ksh[kr * KPAD + seg];
            #pragma unroll
            for (int i = 0; i < 8; ++i)
                *(v8h*)(dst + 8*i) = *(const v8h*)(src + 8*i);
        }
        // ---- stage V^T block: [128 d][64 keys] f16 (pure copy) ---------
        {
            const _Float16* src = Vh + (size_t)tid * S_LEN + kbase;
            _Float16* dst = &VshT[tid * VPAD];
            #pragma unroll
            for (int i = 0; i < 8; ++i)
                *(v8h*)(dst + 8*i) = *(const v8h*)(src + 8*i);
        }
        if (kb + 1 < nkb) {  // prefetch next block
            __builtin_prefetch(Kh + (size_t)(kbase + KBLK + (tid >> 1)) * DHEAD + (tid & 1) * 64, 0, 0);
            __builtin_prefetch(Vh + (size_t)tid * S_LEN + kbase + KBLK, 0, 0);
        }
        __syncthreads();

        // ---- S = Q K^T : four 16x16 score tiles (keys 16t..16t+15) -----
        v8f s[4];
        #pragma unroll
        for (int t = 0; t < 4; ++t) s[t] = (v8f){0,0,0,0,0,0,0,0};
        #pragma unroll
        for (int c = 0; c < 4; ++c) {
            #pragma unroll
            for (int tp = 0; tp < 4; tp += 2) {
                const _Float16* kp0 = &Ksh[(ln + 16* tp   ) * KPAD + 32*c + lh*16];
                const _Float16* kp1 = &Ksh[(ln + 16*(tp+1)) * KPAD + 32*c + lh*16];
                U16h bk0, bk1;
                ((v8h*)&bk0.v)[0] = *(const v8h*)(kp0);
                ((v8h*)&bk0.v)[1] = *(const v8h*)(kp0 + 8);
                ((v8h*)&bk1.v)[0] = *(const v8h*)(kp1);
                ((v8h*)&bk1.v)[1] = *(const v8h*)(kp1 + 8);
                s[tp]   = __builtin_amdgcn_wmma_f32_16x16x32_f16(false, aq[c].v, false, bk0.v, (short)0, s[tp],   false, false);
                s[tp+1] = __builtin_amdgcn_wmma_f32_16x16x32_f16(false, aq[c].v, false, bk1.v, (short)0, s[tp+1], false, false);
            }
        }

        // ---- scale (into log2 domain) ----------------------------------
        #pragma unroll
        for (int t = 0; t < 4; ++t) {
            #pragma unroll
            for (int r = 0; r < 8; ++r)
                s[t][r] *= LSCALE;
        }
        // ---- causal mask: only the diagonal block can be masked --------
        if (kb == nkb - 1) {
            #pragma unroll
            for (int t = 0; t < 4; ++t) {
                const int colt = kbase + 16*t + ln;
                #pragma unroll
                for (int r = 0; r < 8; ++r) {
                    const int qrow = qbase + r + 8*lh;
                    if (colt > qrow) s[t][r] = -1e30f;
                }
            }
        }
        // ---- running max ----------------------------------------------
        float mb[8];
        #pragma unroll
        for (int r = 0; r < 8; ++r)
            mb[r] = fmaxf(fmaxf(s[0][r], s[1][r]), fmaxf(s[2][r], s[3][r]));
        #pragma unroll
        for (int off = 1; off < 16; off <<= 1) {
            #pragma unroll
            for (int r = 0; r < 8; ++r)
                mb[r] = fmaxf(mb[r], __shfl_xor(mb[r], off, 32));
        }
        float al[8];
        #pragma unroll
        for (int r = 0; r < 8; ++r) {
            const float mn = fmaxf(mrow[r], mb[r]);
            al[r] = fast_exp2(mrow[r] - mn);
            mrow[r] = mn;
        }
        // exponentiate scores (2^x, scale already folded in)
        #pragma unroll
        for (int t = 0; t < 4; ++t) {
            #pragma unroll
            for (int r = 0; r < 8; ++r)
                s[t][r] = fast_exp2(s[t][r] - mrow[r]);
        }
        // rescale running accumulators (incl. row-sum tile)
        #pragma unroll
        for (int r = 0; r < 8; ++r) sumacc[r] *= al[r];
        #pragma unroll
        for (int t = 0; t < 8; ++t) {
            #pragma unroll
            for (int r = 0; r < 8; ++r) acc[t][r] *= al[r];
        }

        // ---- transpose P (C layout) -> A layout via per-wave LDS -------
        _Float16* pw = &Psh[wave * QTILE * PPAD];
        #pragma unroll
        for (int t = 0; t < 4; ++t) {
            #pragma unroll
            for (int r = 0; r < 8; ++r)
                pw[(r + 8*lh) * PPAD + 16*t + ln] = (_Float16)s[t][r];
        }
        U16h ap[2];
        {
            const _Float16* pr = pw + ln * PPAD;
            #pragma unroll
            for (int g = 0; g < 2; ++g) {
                ((v8h*)&ap[g].v)[0] = *(const v8h*)(pr + g*32 + lh*8);
                ((v8h*)&ap[g].v)[1] = *(const v8h*)(pr + g*32 + 16 + lh*8);
            }
        }

        // ---- row-sum of P via WMMA against ones ------------------------
        sumacc = __builtin_amdgcn_wmma_f32_16x16x32_f16(false, ap[0].v, false, ones.v, (short)0, sumacc, false, false);
        sumacc = __builtin_amdgcn_wmma_f32_16x16x32_f16(false, ap[1].v, false, ones.v, (short)0, sumacc, false, false);

        // ---- O += P V : 8 output tiles across D ------------------------
        #pragma unroll
        for (int t = 0; t < 8; ++t) {
            const _Float16* vb = &VshT[(t*16 + ln) * VPAD];
            #pragma unroll
            for (int g = 0; g < 2; ++g) {
                U16h bv;
                ((v8h*)&bv.v)[0] = *(const v8h*)(vb + g*32 + lh*8);
                ((v8h*)&bv.v)[1] = *(const v8h*)(vb + g*32 + lh*8 + 16);
                acc[t] = __builtin_amdgcn_wmma_f32_16x16x32_f16(false, ap[g].v, false, bv.v, (short)0, acc[t], false, false);
            }
        }
    }

    // ---- normalize and write Out[B, S, H, D] (write-once: NT stores) ----
    #pragma unroll
    for (int r = 0; r < 8; ++r) {
        const int row = qbase + r + 8*lh;
        const float l = fast_exp2(snk2 - mrow[r]) + sumacc[r];  // sink in denominator
        const float inv = 1.0f / l;
        float* orow = Out + ((size_t)row * NHEAD + head) * DHEAD;
        #pragma unroll
        for (int t = 0; t < 8; ++t)
            __builtin_nontemporal_store(acc[t][r] * inv, orow + t*16 + ln);
    }
}

// ---------------------------------------------------------------------------
extern "C" void kernel_launch(void* const* d_in, const int* in_sizes, int n_in,
                              void* d_out, int out_size, void* d_ws, size_t ws_size,
                              hipStream_t stream) {
    (void)in_sizes; (void)n_in; (void)out_size; (void)ws_size;
    const float* Q     = (const float*)d_in[0];
    const float* K     = (const float*)d_in[1];
    const float* V     = (const float*)d_in[2];
    // d_in[3] = attention_mask (causal, reproduced analytically in-kernel)
    const float* sinks = (const float*)d_in[4];
    float* out = (float*)d_out;

    const size_t kv_elems = (size_t)NKV * S_LEN * DHEAD;   // 1M elems
    _Float16* K16  = (_Float16*)d_ws;                      // 2 MB
    _Float16* V16T = K16 + kv_elems;                       // 2 MB

    // preprocess: K -> f16 (same layout), V -> f16 transposed [NKV][D][S]
    cvt_k_kernel <<<dim3(kv_elems / (256*8)), dim3(256), 0, stream>>>(K, K16);
    cvt_vT_kernel<<<dim3(kv_elems / (256*4)), dim3(256), 0, stream>>>(V, V16T);

    dim3 grid(NHEAD * (S_LEN / QBLK));   // 32 heads * 32 q-blocks = 1024
    dim3 block(WAVES * 32);              // 128 threads = 4 waves
    attn_sink_fa_kernel<<<grid, block, 0, stream>>>(Q, K16, V16T, sinks, out);
}
